// DiaM_75780402970931
// MI455X (gfx1250) — matrix operand order repaired and checked by reading, not compiled, not run on hardware
//
#include <hip/hip_runtime.h>

// Multi-head causal attention forward for MI455X (gfx1250, wave32, WMMA).
// v3: panel staging via GLOBAL_LOAD_ASYNC_TO_LDS_B128 (ASYNCcnt path, no VGPR
// round-trip) with s_wait_asynccnt fencing; bf16 pre-conversion of x and
// weights; fused QKV GEMM (128x64 block tile, 12 WMMAs/k-step per wave);
// fused per-(b,h) attention; 128x64 output GEMM.
// All matrix math via v_wmma_f32_16x16x32_bf16, f32 accumulation.
//
// Workspace (ushort elements): Q | K | V | attnOut | x_bf16 (each 65536x512)
//                              | Wq^T | Wk^T | Wv^T | Wo^T (each 512x512)
// = 5*64MB + 2MB = 322 MB.

#define B_SZ   512
#define N_SEQ  128
#define CH     512
#define NH     8
#define DH     64
#define M_TOT  (B_SZ * N_SEQ)   // 65536 rows

typedef __attribute__((ext_vector_type(16))) __bf16       v16bf;
typedef __attribute__((ext_vector_type(8)))  float        v8f;
typedef __attribute__((ext_vector_type(4)))  unsigned int u32x4;

union FragU { u32x4 u[2]; v16bf v; };

static __device__ __forceinline__ unsigned short f2bf(float f) {
    unsigned int u = __builtin_bit_cast(unsigned int, f);
    unsigned int r = (u + 0x7FFFu + ((u >> 16) & 1u)) >> 16;  // RNE
    return (unsigned short)r;
}

// Async 16B global -> LDS copy (ASYNCcnt-tracked, no VGPR round trip).
// LDS dest address = low 32 bits of the generic pointer (ISA flat->LDS rule).
static __device__ __forceinline__ void async_load_b128(
    const unsigned short* gptr, const unsigned short* lptr)
{
    const unsigned int lds_off = (unsigned int)(unsigned long long)lptr;
    asm volatile("global_load_async_to_lds_b128 %0, %1, off"
                 :: "v"(lds_off), "v"((unsigned long long)gptr)
                 : "memory");
}
static __device__ __forceinline__ void wait_async() {
    asm volatile("s_wait_asynccnt 0x0" ::: "memory");
}

// A fragment (16x32 bf16): lane<16 -> row, K {k0..k0+7, k0+16..k0+23};
// lane>=16 -> same row, K {k0+8..k0+15, k0+24..k0+31}. LDS row-major [row][k],
// stride multiple of 8 elements (16B-aligned b128 loads).
static __device__ __forceinline__ v16bf load_frag_a(const unsigned short* lds,
                                                    int row0, int k0, int stride) {
    const int lane = threadIdx.x & 31;
    const int row  = row0 + (lane & 15);
    const int kb   = k0 + ((lane >> 4) << 3);
    FragU f;
    f.u[0] = *(const u32x4*)(lds + row * stride + kb);
    f.u[1] = *(const u32x4*)(lds + row * stride + kb + 16);
    return f.v;
}

// B fragment (32x16 bf16): lane<16 -> column n, K {k0..k0+7, k0+16..k0+23};
// lane>=16 -> K {k0+8..k0+15, k0+24..k0+31}. LDS transposed [col][k].
static __device__ __forceinline__ v16bf load_frag_b(const unsigned short* lds,
                                                    int col0, int k0, int stride) {
    const int lane = threadIdx.x & 31;
    const int col  = col0 + (lane & 15);
    const int kb   = k0 + ((lane >> 4) << 3);
    FragU f;
    f.u[0] = *(const u32x4*)(lds + col * stride + kb);
    f.u[1] = *(const u32x4*)(lds + col * stride + kb + 16);
    return f.v;
}

static __device__ __forceinline__ v8f wmma_bf16(v16bf a, v16bf b, v8f c) {
    return __builtin_amdgcn_wmma_f32_16x16x32_bf16(false, a, false, b,
                                                   (short)0, c, false, false);
}

// ---------------------------------------------------------------------------
// Prep 1: x f32 -> bf16, 8 elements/thread (one b128 store each).
// ---------------------------------------------------------------------------
__global__ __launch_bounds__(256) void x_prep_kernel(
    const float* __restrict__ X, unsigned short* __restrict__ Xb)
{
    const size_t i = ((size_t)blockIdx.x * 256 + threadIdx.x) * 8;
    u32x4 o;
    #pragma unroll
    for (int p = 0; p < 4; ++p) {
        const float2 v = *(const float2*)(X + i + p * 2);
        o[p] = (unsigned int)f2bf(v.x) | ((unsigned int)f2bf(v.y) << 16);
    }
    *(u32x4*)(Xb + i) = o;
}

// ---------------------------------------------------------------------------
// Prep 2: W[k][n] f32 -> Wt[n][k] bf16, 32x32 tile transpose through LDS.
// blockIdx.z selects which of the 4 weight matrices.
// ---------------------------------------------------------------------------
__global__ __launch_bounds__(256) void w_prep_kernel(
    const float* __restrict__ W0, const float* __restrict__ W1,
    const float* __restrict__ W2, const float* __restrict__ W3,
    unsigned short* __restrict__ T0, unsigned short* __restrict__ T1,
    unsigned short* __restrict__ T2, unsigned short* __restrict__ T3)
{
    const float* W;
    unsigned short* T;
    switch (blockIdx.z) {
        case 0:  W = W0; T = T0; break;
        case 1:  W = W1; T = T1; break;
        case 2:  W = W2; T = T2; break;
        default: W = W3; T = T3; break;
    }
    __shared__ unsigned short tile[32][33];
    const int tid = threadIdx.x;
    const int k0 = blockIdx.y * 32;
    const int n0 = blockIdx.x * 32;
    {
        const int n_l = tid & 31;
        const int kq  = tid >> 5;
        #pragma unroll
        for (int p = 0; p < 4; ++p) {
            const int k_l = kq + p * 8;
            tile[k_l][n_l] = f2bf(W[(size_t)(k0 + k_l) * CH + n0 + n_l]);
        }
    }
    __syncthreads();
    {
        const int k_l = tid & 31;
        const int nq  = tid >> 5;
        #pragma unroll
        for (int p = 0; p < 4; ++p) {
            const int n_l = nq + p * 8;
            T[(size_t)(n0 + n_l) * CH + k0 + k_l] = tile[k_l][n_l];
        }
    }
}

// ---------------------------------------------------------------------------
// Epilogue helper: 32x32 wave tile -> bf16 output with bias.
// ---------------------------------------------------------------------------
static __device__ __forceinline__ void store_tile_bf16(
    const v8f (&acc)[2][2], const float* __restrict__ bias,
    unsigned short* __restrict__ Ob, int m0, int n0, int lane)
{
    const int coll = lane & 15;
    const int rh   = (lane >> 4) << 3;
    #pragma unroll
    for (int ni = 0; ni < 2; ++ni) {
        const int col  = n0 + ni * 16 + coll;
        const float bs = bias[col];
        #pragma unroll
        for (int mi = 0; mi < 2; ++mi) {
            const int rb = m0 + mi * 16 + rh;
            #pragma unroll
            for (int j = 0; j < 8; ++j)
                Ob[(size_t)(rb + j) * CH + col] = f2bf(acc[mi][ni][j] + bs);
        }
    }
}

// ---------------------------------------------------------------------------
// Fused QKV GEMM: {Q,K,V}_bf16 = x_bf16 @ W{q,k,v}^T(+bias). Block tile
// 128Mx64N, 8 waves as 4(M)x2(N), each wave 32x32 per output => 12 WMMAs per
// k-step reusing one staged x panel across all three weights. Panels staged
// with async global->LDS b128 copies.
// ---------------------------------------------------------------------------
__global__ __launch_bounds__(256) void qkv_gemm_kernel(
    const unsigned short* __restrict__ Xb,
    const unsigned short* __restrict__ Wqt, const unsigned short* __restrict__ Wkt,
    const unsigned short* __restrict__ Wvt,
    const float* __restrict__ bq, const float* __restrict__ bk,
    const float* __restrict__ bv,
    unsigned short* __restrict__ Qb, unsigned short* __restrict__ Kb,
    unsigned short* __restrict__ Vb)
{
    __shared__ __align__(16) unsigned short As[128 * 72];  // x panel [m][k]
    __shared__ __align__(16) unsigned short Bq[64 * 72];   // Wq^T panel [n][k]
    __shared__ __align__(16) unsigned short Bk[64 * 72];
    __shared__ __align__(16) unsigned short Bv[64 * 72];

    const int tid    = threadIdx.x;
    const int m_base = blockIdx.y * 128;
    const int n_base = blockIdx.x * 64;
    const int wave   = tid >> 5;
    const int lane   = tid & 31;
    const int m_off  = (wave & 3) * 32;
    const int n_off  = (wave >> 2) * 32;

    v8f aq[2][2], ak[2][2], av[2][2];
    #pragma unroll
    for (int mi = 0; mi < 2; ++mi)
        #pragma unroll
        for (int ni = 0; ni < 2; ++ni)
            #pragma unroll
            for (int i = 0; i < 8; ++i) {
                aq[mi][ni][i] = 0.0f; ak[mi][ni][i] = 0.0f; av[mi][ni][i] = 0.0f;
            }

    for (int kb = 0; kb < CH; kb += 64) {
        __syncthreads();  // previous iteration's readers are done
        {   // x panel: 128x64 bf16 = 1024 16B chunks, 4/thread, async
            #pragma unroll
            for (int p = 0; p < 4; ++p) {
                const int c  = tid + p * 256;
                const int r  = c >> 3;
                const int k8 = (c & 7) * 8;
                async_load_b128(Xb + (size_t)(m_base + r) * CH + kb + k8,
                                As + r * 72 + k8);
            }
            // three 64x64 bf16 weight panels = 512 chunks each, 2/thread
            #pragma unroll
            for (int p = 0; p < 2; ++p) {
                const int c  = tid + p * 256;
                const int n  = c >> 3;
                const int k8 = (c & 7) * 8;
                const size_t g = (size_t)(n_base + n) * CH + kb + k8;
                const int    l = n * 72 + k8;
                async_load_b128(Wqt + g, Bq + l);
                async_load_b128(Wkt + g, Bk + l);
                async_load_b128(Wvt + g, Bv + l);
            }
            wait_async();
        }
        __syncthreads();
        #pragma unroll
        for (int ks = 0; ks < 64; ks += 32) {
            const v16bf a0 = load_frag_a(As, m_off, ks, 72);
            const v16bf a1 = load_frag_a(As, m_off + 16, ks, 72);
            {
                const v16bf b0 = load_frag_b(Bq, n_off, ks, 72);
                const v16bf b1 = load_frag_b(Bq, n_off + 16, ks, 72);
                aq[0][0] = wmma_bf16(a0, b0, aq[0][0]);
                aq[0][1] = wmma_bf16(a0, b1, aq[0][1]);
                aq[1][0] = wmma_bf16(a1, b0, aq[1][0]);
                aq[1][1] = wmma_bf16(a1, b1, aq[1][1]);
            }
            {
                const v16bf b0 = load_frag_b(Bk, n_off, ks, 72);
                const v16bf b1 = load_frag_b(Bk, n_off + 16, ks, 72);
                ak[0][0] = wmma_bf16(a0, b0, ak[0][0]);
                ak[0][1] = wmma_bf16(a0, b1, ak[0][1]);
                ak[1][0] = wmma_bf16(a1, b0, ak[1][0]);
                ak[1][1] = wmma_bf16(a1, b1, ak[1][1]);
            }
            {
                const v16bf b0 = load_frag_b(Bv, n_off, ks, 72);
                const v16bf b1 = load_frag_b(Bv, n_off + 16, ks, 72);
                av[0][0] = wmma_bf16(a0, b0, av[0][0]);
                av[0][1] = wmma_bf16(a0, b1, av[0][1]);
                av[1][0] = wmma_bf16(a1, b0, av[1][0]);
                av[1][1] = wmma_bf16(a1, b1, av[1][1]);
            }
        }
    }
    const int m0 = m_base + m_off;
    const int n0 = n_base + n_off;
    store_tile_bf16(aq, bq, Qb, m0, n0, lane);
    store_tile_bf16(ak, bk, Kb, m0, n0, lane);
    store_tile_bf16(av, bv, Vb, m0, n0, lane);
}

// ---------------------------------------------------------------------------
// Output GEMM: out_f32 = attn_bf16 @ Wo^T + bo. 128x64 block tile, 32x32/wave.
// ---------------------------------------------------------------------------
__global__ __launch_bounds__(256) void out_gemm_kernel(
    const unsigned short* __restrict__ Ain, const unsigned short* __restrict__ Wt,
    const float* __restrict__ bias, float* __restrict__ O)
{
    __shared__ __align__(16) unsigned short As[128 * 72];
    __shared__ __align__(16) unsigned short Bs[64 * 72];
    const int tid    = threadIdx.x;
    const int m_base = blockIdx.y * 128;
    const int n_base = blockIdx.x * 64;
    const int wave   = tid >> 5;
    const int lane   = tid & 31;
    const int m_off  = (wave & 3) * 32;
    const int n_off  = (wave >> 2) * 32;

    v8f acc[2][2];
    #pragma unroll
    for (int mi = 0; mi < 2; ++mi)
        #pragma unroll
        for (int ni = 0; ni < 2; ++ni)
            #pragma unroll
            for (int i = 0; i < 8; ++i) acc[mi][ni][i] = 0.0f;

    for (int kb = 0; kb < CH; kb += 64) {
        __syncthreads();
        {
            #pragma unroll
            for (int p = 0; p < 4; ++p) {
                const int c  = tid + p * 256;
                const int r  = c >> 3;
                const int k8 = (c & 7) * 8;
                async_load_b128(Ain + (size_t)(m_base + r) * CH + kb + k8,
                                As + r * 72 + k8);
            }
            #pragma unroll
            for (int p = 0; p < 2; ++p) {
                const int c  = tid + p * 256;
                const int n  = c >> 3;
                const int k8 = (c & 7) * 8;
                async_load_b128(Wt + (size_t)(n_base + n) * CH + kb + k8,
                                Bs + n * 72 + k8);
            }
            wait_async();
        }
        __syncthreads();
        #pragma unroll
        for (int ks = 0; ks < 64; ks += 32) {
            const v16bf a0 = load_frag_a(As, m_off, ks, 72);
            const v16bf a1 = load_frag_a(As, m_off + 16, ks, 72);
            const v16bf b0 = load_frag_b(Bs, n_off, ks, 72);
            const v16bf b1 = load_frag_b(Bs, n_off + 16, ks, 72);
            acc[0][0] = wmma_bf16(a0, b0, acc[0][0]);
            acc[0][1] = wmma_bf16(a0, b1, acc[0][1]);
            acc[1][0] = wmma_bf16(a1, b0, acc[1][0]);
            acc[1][1] = wmma_bf16(a1, b1, acc[1][1]);
        }
    }
    const int coll = lane & 15;
    const int rh   = (lane >> 4) << 3;
    #pragma unroll
    for (int ni = 0; ni < 2; ++ni) {
        const int col  = n_base + n_off + ni * 16 + coll;
        const float bs = bias[col];
        #pragma unroll
        for (int mi = 0; mi < 2; ++mi) {
            const int rb = m_base + m_off + mi * 16 + rh;
            #pragma unroll
            for (int j = 0; j < 8; ++j)
                O[(size_t)(rb + j) * CH + col] = acc[mi][ni][j] + bs;
        }
    }
}

// ---------------------------------------------------------------------------
// Fused causal attention per (batch, head). Q,K,V bf16 tiles [128,64].
// Q/K staged async to LDS; V transposed through VGPRs. Phase 1: S = Q K^T
// (16 WMMAs/wave) -> mask+scale -> register softmax (16-lane __shfl_xor
// reductions; lane halves hold disjoint rows). Phase 2: O = P V against
// LDS-transposed V. S overlays Q/K LDS region.
// ---------------------------------------------------------------------------
__global__ __launch_bounds__(256) void attn_kernel(
    const unsigned short* __restrict__ Qb, const unsigned short* __restrict__ Kb,
    const unsigned short* __restrict__ Vb, unsigned short* __restrict__ Ab)
{
    __shared__ __align__(16) unsigned short SQK[2 * 128 * 72];  // Q|K, reused for S
    __shared__ __align__(16) unsigned short Vt[64 * 136];       // V^T [d][key]
    unsigned short* Qs = SQK;             // [128][72]
    unsigned short* Ks = SQK + 128 * 72;  // [128][72]
    unsigned short* Ss = SQK;             // [128][136] (after phase 1)

    const int tid    = threadIdx.x;
    const int bh     = blockIdx.x;
    const int bat    = bh >> 3;
    const int head   = bh & 7;
    const size_t r0g = (size_t)bat * N_SEQ;
    const int cbase  = head * DH;

    {   // stage Q,K (async b128) and V^T (element transpose)
        #pragma unroll
        for (int p = 0; p < 4; ++p) {
            const int c  = tid + p * 256;
            const int r  = c >> 3;
            const int d8 = (c & 7) * 8;
            const size_t g = (r0g + r) * CH + cbase + d8;
            async_load_b128(Qb + g, Qs + r * 72 + d8);
            async_load_b128(Kb + g, Ks + r * 72 + d8);
        }
        const int d2 = (tid & 31) * 2;
        const int r0 = tid >> 5;
        #pragma unroll
        for (int p = 0; p < 16; ++p) {
            const int r = r0 + p * 8;
            const unsigned int v =
                *(const unsigned int*)(Vb + (r0g + r) * CH + cbase + d2);
            Vt[d2 * 136 + r]       = (unsigned short)(v & 0xFFFFu);
            Vt[(d2 + 1) * 136 + r] = (unsigned short)(v >> 16);
        }
        wait_async();
    }
    __syncthreads();

    const int wave  = tid >> 5;
    const int lane  = tid & 31;
    const int q0    = wave * 16;          // query rows q0..q0+15
    const int rhalf = (lane >> 4) << 3;
    const int coll  = lane & 15;

    v8f acc[8];
    #pragma unroll
    for (int t = 0; t < 8; ++t)
        #pragma unroll
        for (int i = 0; i < 8; ++i) acc[t][i] = 0.0f;

    #pragma unroll
    for (int ks = 0; ks < DH; ks += 32) {
        const v16bf a = load_frag_a(Qs, q0, ks, 72);
        #pragma unroll
        for (int t = 0; t < 8; ++t) {
            const v16bf b = load_frag_b(Ks, t * 16, ks, 72);
            acc[t] = wmma_bf16(a, b, acc[t]);
        }
    }

    // causal mask (added pre-scale), scale by 1/sqrt(64), row softmax
    #pragma unroll
    for (int j = 0; j < 8; ++j) {
        const int q = q0 + j + rhalf;
        float m = -3.0e38f;
        #pragma unroll
        for (int t = 0; t < 8; ++t) {
            const int col = t * 16 + coll;
            float s = acc[t][j];
            s = (col <= q) ? s * 0.125f : (s - 100000.0f) * 0.125f;
            acc[t][j] = s;
            m = fmaxf(m, s);
        }
        #pragma unroll
        for (int off = 1; off < 16; off <<= 1)
            m = fmaxf(m, __shfl_xor(m, off, 32));
        float sum = 0.0f;
        #pragma unroll
        for (int t = 0; t < 8; ++t) {
            const float p = __expf(acc[t][j] - m);
            acc[t][j] = p;
            sum += p;
        }
        #pragma unroll
        for (int off = 1; off < 16; off <<= 1)
            sum += __shfl_xor(sum, off, 32);
        const float inv = 1.0f / sum;
        #pragma unroll
        for (int t = 0; t < 8; ++t) acc[t][j] *= inv;
    }

    __syncthreads();  // all waves done reading Qs/Ks
    #pragma unroll
    for (int j = 0; j < 8; ++j) {
        const int q = q0 + j + rhalf;
        #pragma unroll
        for (int t = 0; t < 8; ++t)
            Ss[q * 136 + t * 16 + coll] = f2bf(acc[t][j]);
    }
    __syncthreads();

    v8f oacc[4];
    #pragma unroll
    for (int t = 0; t < 4; ++t)
        #pragma unroll
        for (int i = 0; i < 8; ++i) oacc[t][i] = 0.0f;

    #pragma unroll
    for (int ks = 0; ks < N_SEQ; ks += 32) {
        const v16bf a = load_frag_a(Ss, q0, ks, 136);
        #pragma unroll
        for (int t = 0; t < 4; ++t) {
            const v16bf b = load_frag_b(Vt, t * 16, ks, 136);
            oacc[t] = wmma_bf16(a, b, oacc[t]);
        }
    }

    #pragma unroll
    for (int t = 0; t < 4; ++t) {
        const int c = cbase + t * 16 + coll;
        #pragma unroll
        for (int j = 0; j < 8; ++j) {
            const int q = q0 + j + rhalf;
            Ab[(r0g + q) * CH + c] = f2bf(oacc[t][j]);
        }
    }
}

// ---------------------------------------------------------------------------
extern "C" void kernel_launch(void* const* d_in, const int* in_sizes, int n_in,
                              void* d_out, int out_size, void* d_ws, size_t ws_size,
                              hipStream_t stream) {
    const float* x  = (const float*)d_in[0];
    const float* Wq = (const float*)d_in[1];
    const float* bq = (const float*)d_in[2];
    const float* Wk = (const float*)d_in[3];
    const float* bk = (const float*)d_in[4];
    const float* Wv = (const float*)d_in[5];
    const float* bv = (const float*)d_in[6];
    const float* Wo = (const float*)d_in[7];
    const float* bo = (const float*)d_in[8];
    float* out = (float*)d_out;

    const size_t per = (size_t)M_TOT * CH;      // 33.5M elements
    const size_t wsz = (size_t)CH * CH;         // 262144 elements
    unsigned short* Qb  = (unsigned short*)d_ws;
    unsigned short* Kb  = Qb + per;
    unsigned short* Vb  = Kb + per;
    unsigned short* Ab  = Vb + per;
    unsigned short* Xb  = Ab + per;
    unsigned short* Wqt = Xb + per;
    unsigned short* Wkt = Wqt + wsz;
    unsigned short* Wvt = Wkt + wsz;
    unsigned short* Wot = Wvt + wsz;

    dim3 bb(256);
    x_prep_kernel<<<dim3((unsigned)(per / (256 * 8))), bb, 0, stream>>>(x, Xb);
    w_prep_kernel<<<dim3(CH / 32, CH / 32, 4), bb, 0, stream>>>(
        Wq, Wk, Wv, Wo, Wqt, Wkt, Wvt, Wot);
    qkv_gemm_kernel<<<dim3(CH / 64, M_TOT / 128), bb, 0, stream>>>(
        Xb, Wqt, Wkt, Wvt, bq, bk, bv, Qb, Kb, Vb);
    attn_kernel<<<dim3(B_SZ * NH), bb, 0, stream>>>(Qb, Kb, Vb, Ab);
    out_gemm_kernel<<<dim3(CH / 64, M_TOT / 128), bb, 0, stream>>>(Ab, Wot, bo, out);
}
